// Attention_24988119728870
// MI455X (gfx1250) — compile-verified, matrix operand order checked by reading, not run on hardware
//
#include <hip/hip_runtime.h>
#include <math.h>

// ---------------------------------------------------------------- constants
#define DIM   4096
#define NH    32
#define HD    128
#define BSZ   2
#define SEQ   2048
#define ALEN  10
#define NTOK  (BSZ * SEQ)

typedef __bf16 bf16_t;
typedef __attribute__((ext_vector_type(16))) __bf16 v16bf;
typedef __attribute__((ext_vector_type(8)))  float  v8f;

union FragBF { uint4 u4[2]; v16bf v; };

__device__ __forceinline__ v8f wmma_bf16f32(v16bf a, v16bf b, v8f c) {
  // D = A(16x32 bf16) * B(32x16 bf16) + C(16x16 f32)
  return __builtin_amdgcn_wmma_f32_16x16x32_bf16(
      /*neg_a=*/false, a, /*neg_b=*/false, b,
      /*c_mod=*/(short)0, c, /*reuse_a=*/false, /*reuse_b=*/false);
}

// ---- CDNA5 async Global->LDS (ASYNCcnt-tracked, bypasses VGPRs) ----------
__device__ __forceinline__ void async_load_b128(unsigned lds_off, const void* gaddr) {
  // VDST = VGPR holding LDS byte address; VADDR = 64-bit global address.
  asm volatile("global_load_async_to_lds_b128 %0, %1, off"
               :: "v"(lds_off), "v"(gaddr) : "memory");
}
__device__ __forceinline__ void wait_async0() {
  asm volatile("s_wait_asynccnt 0x0" ::: "memory");
}
__device__ __forceinline__ unsigned lds_off_u32(const void* p) {
  // Generic->LDS: the low 32 bits of the generic address are the LDS byte offset.
  return (unsigned)(size_t)p;
}

// ---------------------------------------------------------------- f32 -> bf16
__global__ void cvt_f32_bf16(const float* __restrict__ src,
                             bf16_t* __restrict__ dst, size_t n) {
  size_t i = (size_t)blockIdx.x * blockDim.x + threadIdx.x;
  if (i < n) dst[i] = (bf16_t)src[i];
}

// ---------------------------------------------------------------- WMMA GEMM
// C[M,N] (f32) = A[M,K](bf16, row-major) @ B[N,K](bf16, row-major)^T
// Double-buffered LDS, async global->LDS staging overlapped with WMMA.
#define BM 128
#define BN 64
#define BK 32
#define A_STRIDE 40   // bf16 elems per LDS row (32 data + 8 pad, 16B aligned)
#define B_STRIDE 40

__global__ __launch_bounds__(256)
void gemm_bf16(const bf16_t* __restrict__ A, const bf16_t* __restrict__ B,
               float* __restrict__ C, int M, int N, int K) {
  __shared__ __align__(16) bf16_t As[2][BM * A_STRIDE];  // 2 x 10 KB
  __shared__ __align__(16) bf16_t Bs[2][BN * B_STRIDE];  // 2 x  5 KB

  const int tid  = threadIdx.x;
  const int wave = tid >> 5;
  const int lane = tid & 31;
  const int laneLo = lane & 15, laneHalf = lane >> 4;
  const int waveM = wave >> 1;        // 0..3
  const int waveN = wave & 1;         // 0..1
  const int blockM = blockIdx.y * BM;
  const int blockN = blockIdx.x * BN;

  const unsigned AsBase = lds_off_u32(&As[0][0]);
  const unsigned BsBase = lds_off_u32(&Bs[0][0]);
  const unsigned ASZ = BM * A_STRIDE * 2;   // bytes per A buffer
  const unsigned BSZ_ = BN * B_STRIDE * 2;  // bytes per B buffer

  // per-thread staging slices (3 async b128 per thread per tile)
  const int arow0 = tid >> 1, acol0 = (tid & 1) * 2;   // rows 0..127, uint4 {0,1}/{2,3}
  const int brow  = tid >> 2, bcol  = tid & 3;         // rows 0..63

  auto stage = [&](int buf, int k0) {
    const unsigned ab = AsBase + (unsigned)buf * ASZ;
    const unsigned bb = BsBase + (unsigned)buf * BSZ_;
    // A tile: 2 x b128 per thread (rows only feed discarded C rows if >= M)
    async_load_b128(ab + (unsigned)(arow0 * 5 + acol0) * 16,
                    A + (size_t)(blockM + arow0) * K + k0 + acol0 * 8);
    async_load_b128(ab + (unsigned)(arow0 * 5 + acol0 + 1) * 16,
                    A + (size_t)(blockM + arow0) * K + k0 + acol0 * 8 + 8);
    // B tile: 1 x b128 per thread
    async_load_b128(bb + (unsigned)(brow * 5 + bcol) * 16,
                    B + (size_t)(blockN + brow) * K + k0 + bcol * 8);
  };

  v8f acc[2][2] = {};
  const int nk = K / BK;

  stage(0, 0);                                   // prologue
  int buf = 0;
  for (int kt = 0; kt < nk; ++kt) {
    wait_async0();                               // tile kt landed (this wave's slice)
    __syncthreads();                             // all slices landed / readers done
    if (kt + 1 < nk) stage(buf ^ 1, (kt + 1) * BK);  // stream next tile during compute
    if (kt + 2 < nk) {                           // keep GL2 warm two tiles ahead
      __builtin_prefetch(A + (size_t)(blockM + arow0) * K + (kt + 2) * BK, 0, 1);
      __builtin_prefetch(B + (size_t)(blockN + brow) * K + (kt + 2) * BK, 0, 1);
    }

    // ---- fragments (16-bit A-matrix layout: per-lane 2x b128)
    FragBF a[2], b[2];
    #pragma unroll
    for (int i = 0; i < 2; ++i) {
      int row = waveM * 32 + i * 16 + laneLo;
      const uint4* p = (const uint4*)&As[buf][0] + row * 5;
      a[i].u4[0] = p[laneHalf];
      a[i].u4[1] = p[2 + laneHalf];
    }
    #pragma unroll
    for (int j = 0; j < 2; ++j) {
      int row = waveN * 32 + j * 16 + laneLo;
      const uint4* p = (const uint4*)&Bs[buf][0] + row * 5;
      b[j].u4[0] = p[laneHalf];
      b[j].u4[1] = p[2 + laneHalf];
    }
    #pragma unroll
    for (int i = 0; i < 2; ++i)
      #pragma unroll
      for (int j = 0; j < 2; ++j)
        acc[i][j] = wmma_bf16f32(a[i].v, b[j].v, acc[i][j]);
    buf ^= 1;
  }

  // ---- store C (f32 16x16 C/D layout: VGPR r -> row r (+8 for upper half))
  #pragma unroll
  for (int i = 0; i < 2; ++i)
    #pragma unroll
    for (int j = 0; j < 2; ++j) {
      int n = blockN + waveN * 32 + j * 16 + laneLo;
      #pragma unroll
      for (int r = 0; r < 8; ++r) {
        int m = blockM + waveM * 32 + i * 16 + r + laneHalf * 8;
        if (m < M) C[(size_t)m * N + n] = acc[i][j][r];
      }
    }
}

// ---------------------------------------------------------------- RoPE (Q,K)
// xq/xk: f32 [b, s, h, d].  Out: bf16 [b, h, s, d].
__global__ void rope_qk(const float* __restrict__ xq, const float* __restrict__ xk,
                        const float* __restrict__ cosT, const float* __restrict__ sinT,
                        bf16_t* __restrict__ Qb, bf16_t* __restrict__ Kb) {
  size_t idx = (size_t)blockIdx.x * blockDim.x + threadIdx.x;  // (b,s,h,i<64)
  size_t total = (size_t)BSZ * SEQ * NH * (HD / 2);
  if (idx >= total) return;
  int i = idx & 63;
  size_t t = idx >> 6;
  int h = (int)(t % NH); t /= NH;
  int s = (int)(t % SEQ);
  int b = (int)(t / SEQ);
  float c  = cosT[(size_t)s * 64 + i];
  float sn = sinT[(size_t)s * 64 + i];
  size_t src = ((size_t)(b * SEQ + s)) * DIM + h * HD + 2 * i;
  size_t dst = (((size_t)(b * NH + h)) * SEQ + s) * HD + 2 * i;
  float qr = xq[src], qi = xq[src + 1];
  Qb[dst]     = (bf16_t)(qr * c - qi * sn);
  Qb[dst + 1] = (bf16_t)(qr * sn + qi * c);
  float kr = xk[src], ki = xk[src + 1];
  Kb[dst]     = (bf16_t)(kr * c - ki * sn);
  Kb[dst + 1] = (bf16_t)(kr * sn + ki * c);
}

// ---------------------------------------------------------------- V transpose
// xv: f32 [b, s, h, d]  ->  Vt: bf16 [b, h, d, s]   (LDS-tiled, coalesced)
__global__ void v_transpose(const float* __restrict__ xv, bf16_t* __restrict__ Vt) {
  __shared__ float tile[32][33];
  int bh = blockIdx.z;
  int b = bh / NH, h = bh % NH;
  int s0 = blockIdx.x * 32, d0 = blockIdx.y * 32;
  int tx = threadIdx.x, ty = threadIdx.y;          // 32 x 8
  #pragma unroll
  for (int r = 0; r < 32; r += 8) {
    int s = s0 + ty + r, d = d0 + tx;
    tile[ty + r][tx] = xv[((size_t)(b * SEQ + s)) * DIM + h * HD + d];
  }
  __syncthreads();
  #pragma unroll
  for (int r = 0; r < 32; r += 8) {
    int d = d0 + ty + r, s = s0 + tx;
    Vt[(((size_t)(b * NH + h)) * HD + d) * SEQ + s] = (bf16_t)tile[tx][ty + r];
  }
}

// ---------------------------------------------------------------- flash attn
// Q,K: bf16 [b,h,s,d]; Vt: bf16 [b,h,d,s]; Out: f32 [b,s,h*d]. Causal.
#define FA_BQ 64
#define FA_BK 64
#define K_STRIDE 136   // 128 + 8 pad (bf16 elems)
#define V_STRIDE 72    //  64 + 8 pad
#define P_STRIDE 72

__global__ __launch_bounds__(128)
void flash_attn(const bf16_t* __restrict__ Q, const bf16_t* __restrict__ Kc,
                const bf16_t* __restrict__ Vt, float* __restrict__ Out) {
  __shared__ __align__(16) bf16_t Ks[FA_BK * K_STRIDE];   // ~17 KB  [key][d]
  __shared__ __align__(16) bf16_t Vs[HD * V_STRIDE];      // ~18 KB  [d][key]
  __shared__ __align__(16) bf16_t Ps[4 * 16 * P_STRIDE];  // ~9 KB   per-wave P

  const int tid = threadIdx.x, wave = tid >> 5, lane = tid & 31;
  const int laneLo = lane & 15, laneHalf = lane >> 4;
  const int qblk = blockIdx.x;            // 0..SEQ/64-1
  const int h = blockIdx.y, b = blockIdx.z;
  const size_t bh = (size_t)(b * NH + h);
  const bf16_t* Qp = Q  + bh * SEQ * HD;
  const bf16_t* Kp = Kc + bh * SEQ * HD;
  const bf16_t* Vp = Vt + bh * HD * SEQ;
  const int qbase = qblk * FA_BQ + wave * 16;
  const unsigned KsBase = lds_off_u32(Ks);
  const unsigned VsBase = lds_off_u32(Vs);

  // ---- preload Q fragments (16 rows x 128 d = 4 A-fragments)
  FragBF qf[4];
  {
    const uint4* qrow = (const uint4*)(Qp + (size_t)(qbase + laneLo) * HD);
    #pragma unroll
    for (int kk = 0; kk < 4; ++kk) {
      qf[kk].u4[0] = qrow[kk * 4 + laneHalf];
      qf[kk].u4[1] = qrow[kk * 4 + 2 + laneHalf];
    }
  }

  v8f o[8] = {};
  float mrun[8], lrun[8];
  #pragma unroll
  for (int r = 0; r < 8; ++r) { mrun[r] = -__builtin_inff(); lrun[r] = 0.f; }
  const float scale = 0.08838834764831845f;   // 1/sqrt(128)

  const int ntiles = qblk + 1;                // causal: skip fully-masked tiles
  for (int kt = 0; kt < ntiles; ++kt) {
    const int kbase = kt * FA_BK;
    __syncthreads();                          // prev-tile readers done
    // ---- async-stage K tile (64 keys x 128 d) : 8 b128 per thread
    {
      const uint4* src = (const uint4*)(Kp + (size_t)kbase * HD);
      #pragma unroll
      for (int it = 0; it < 8; ++it) {
        int idx = tid + it * 128;             // 0..1023
        int row = idx >> 4, c = idx & 15;
        async_load_b128(KsBase + (unsigned)(row * 17 + c) * 16, src + row * 16 + c);
      }
      // ---- async-stage Vt tile (128 d x 64 keys) : 8 b128 per thread
      #pragma unroll
      for (int it = 0; it < 8; ++it) {
        int idx = tid + it * 128;             // 0..1023
        int d = idx >> 3, c = idx & 7;
        async_load_b128(VsBase + (unsigned)(d * 9 + c) * 16,
                        Vp + (size_t)d * SEQ + kbase + c * 8);
      }
    }
    wait_async0();
    __syncthreads();

    // ---- S = Q K^T (4 N-tiles x 4 K-steps)
    v8f st[4];
    #pragma unroll
    for (int j = 0; j < 4; ++j) {
      v8f accS = {};
      const uint4* p = (const uint4*)Ks + (j * 16 + laneLo) * 17;
      #pragma unroll
      for (int kk = 0; kk < 4; ++kk) {
        FragBF kf;
        kf.u4[0] = p[kk * 4 + laneHalf];
        kf.u4[1] = p[kk * 4 + 2 + laneHalf];
        accS = wmma_bf16f32(qf[kk].v, kf.v, accS);
      }
      st[j] = accS;
    }

    // ---- online softmax (rows spread over 16 lanes of each half-wave)
    #pragma unroll
    for (int r = 0; r < 8; ++r) {
      const int q = qbase + r + laneHalf * 8;
      float mx = -__builtin_inff();
      #pragma unroll
      for (int j = 0; j < 4; ++j) {
        int kidx = kbase + j * 16 + laneLo;
        float v = st[j][r] * scale + (kidx > q ? -1.0e9f : 0.f);
        st[j][r] = v;
        mx = fmaxf(mx, v);
      }
      #pragma unroll
      for (int off = 8; off >= 1; off >>= 1)
        mx = fmaxf(mx, __shfl_xor(mx, off, 32));
      float nm = fmaxf(mrun[r], mx);
      float alpha = __expf(mrun[r] - nm);
      mrun[r] = nm;
      float rs = 0.f;
      #pragma unroll
      for (int j = 0; j < 4; ++j) {
        float p = __expf(st[j][r] - nm);
        st[j][r] = p;
        rs += p;
      }
      #pragma unroll
      for (int off = 8; off >= 1; off >>= 1)
        rs += __shfl_xor(rs, off, 32);
      lrun[r] = lrun[r] * alpha + rs;
      #pragma unroll
      for (int dt = 0; dt < 8; ++dt) o[dt][r] *= alpha;
    }

    // ---- relayout P (C-frag) -> A-frag via per-wave LDS strip
    bf16_t* pb = Ps + wave * 16 * P_STRIDE;
    __builtin_amdgcn_wave_barrier();
    #pragma unroll
    for (int j = 0; j < 4; ++j)
      #pragma unroll
      for (int r = 0; r < 8; ++r)
        pb[(r + laneHalf * 8) * P_STRIDE + j * 16 + laneLo] = (bf16_t)st[j][r];
    __builtin_amdgcn_wave_barrier();

    FragBF pf[2];
    {
      const uint4* p = (const uint4*)(pb + (size_t)laneLo * P_STRIDE);
      #pragma unroll
      for (int k2 = 0; k2 < 2; ++k2) {
        pf[k2].u4[0] = p[k2 * 4 + laneHalf];
        pf[k2].u4[1] = p[k2 * 4 + 2 + laneHalf];
      }
    }

    // ---- O += P @ V  (8 d-tiles x 2 K-steps)
    #pragma unroll
    for (int dt = 0; dt < 8; ++dt) {
      const uint4* p = (const uint4*)Vs + (dt * 16 + laneLo) * 9;
      #pragma unroll
      for (int k2 = 0; k2 < 2; ++k2) {
        FragBF vf;
        vf.u4[0] = p[k2 * 4 + laneHalf];
        vf.u4[1] = p[k2 * 4 + 2 + laneHalf];
        o[dt] = wmma_bf16f32(pf[k2].v, vf.v, o[dt]);
      }
    }
  }

  // ---- epilogue: O /= l, store f32 [b, s, h*d]
  #pragma unroll
  for (int dt = 0; dt < 8; ++dt)
    #pragma unroll
    for (int r = 0; r < 8; ++r) {
      int q = qbase + r + laneHalf * 8;
      Out[((size_t)(b * SEQ + q)) * DIM + h * HD + dt * 16 + laneLo] =
          o[dt][r] / lrun[r];
    }
}

// ---------------------------------------------------------------- adapter attn
// Out[b,s,h*d] += gate[h] * softmax(Q . ak^T * scale) @ av   (one wave per b,h,q)
__global__ __launch_bounds__(256)
void adapter_attn(const bf16_t* __restrict__ Q, const float* __restrict__ akf,
                  const float* __restrict__ avf, const float* __restrict__ gate,
                  float* __restrict__ Out) {
  int gw   = blockIdx.x * (blockDim.x >> 5) + (threadIdx.x >> 5);
  int lane = threadIdx.x & 31;
  int q  = gw % SEQ;
  int bh = gw / SEQ;
  int h  = bh % NH;
  int b  = bh / NH;
  if (b >= BSZ) return;

  const bf16_t* qp = Q + (((size_t)(b * NH + h)) * SEQ + q) * HD;
  float qv[4];
  #pragma unroll
  for (int i = 0; i < 4; ++i) qv[i] = (float)qp[lane * 4 + i];

  const float scale = 0.08838834764831845f;
  float s[ALEN];
  #pragma unroll
  for (int a = 0; a < ALEN; ++a) {
    const float* ak = akf + (size_t)a * DIM + h * HD;
    float part = 0.f;
    #pragma unroll
    for (int i = 0; i < 4; ++i) part += qv[i] * ak[lane * 4 + i];
    #pragma unroll
    for (int off = 16; off >= 1; off >>= 1) part += __shfl_xor(part, off, 32);
    s[a] = part * scale;
  }
  float mx = s[0];
  #pragma unroll
  for (int a = 1; a < ALEN; ++a) mx = fmaxf(mx, s[a]);
  float sum = 0.f;
  #pragma unroll
  for (int a = 0; a < ALEN; ++a) { s[a] = __expf(s[a] - mx); sum += s[a]; }
  float g = gate[h] / sum;

  float accd[4] = {0.f, 0.f, 0.f, 0.f};
  #pragma unroll
  for (int a = 0; a < ALEN; ++a) {
    const float* av = avf + (size_t)a * DIM + h * HD;
    float c = s[a] * g;
    #pragma unroll
    for (int i = 0; i < 4; ++i) accd[i] += c * av[lane * 4 + i];
  }
  float* op = Out + ((size_t)(b * SEQ + q)) * DIM + h * HD;
  #pragma unroll
  for (int i = 0; i < 4; ++i) op[lane * 4 + i] += accd[i];
}

// ---------------------------------------------------------------- launch
extern "C" void kernel_launch(void* const* d_in, const int* in_sizes, int n_in,
                              void* d_out, int out_size, void* d_ws, size_t ws_size,
                              hipStream_t stream) {
  const float* x       = (const float*)d_in[0];
  const float* adapter = (const float*)d_in[1];
  const float* wq      = (const float*)d_in[2];
  const float* wk      = (const float*)d_in[3];
  const float* wv      = (const float*)d_in[4];
  const float* wo      = (const float*)d_in[5];
  const float* gate    = (const float*)d_in[6];
  const float* fcos    = (const float*)d_in[7];
  const float* fsin    = (const float*)d_in[8];
  float* out = (float*)d_out;

  // -------- workspace layout (256B-aligned regions)
  char* base = (char*)d_ws;
  size_t off = 0;
  auto alloc = [&](size_t bytes) {
    size_t o = off;
    off += (bytes + 255) & ~(size_t)255;
    return o;
  };
  const size_t sz_tok_bf = (size_t)NTOK * DIM * sizeof(bf16_t);
  const size_t sz_tok_f  = (size_t)NTOK * DIM * sizeof(float);
  const size_t sz_w_bf   = (size_t)DIM * DIM * sizeof(bf16_t);

  bf16_t* x_bf  = (bf16_t*)(base + alloc(sz_tok_bf));
  bf16_t* wq_bf = (bf16_t*)(base + alloc(sz_w_bf));
  bf16_t* wk_bf = (bf16_t*)(base + alloc(sz_w_bf));
  bf16_t* wv_bf = (bf16_t*)(base + alloc(sz_w_bf));
  bf16_t* wo_bf = (bf16_t*)(base + alloc(sz_w_bf));
  bf16_t* ad_bf = (bf16_t*)(base + alloc((size_t)ALEN * DIM * sizeof(bf16_t)));
  float*  xq_f  = (float*)(base + alloc(sz_tok_f));
  float*  xk_f  = (float*)(base + alloc(sz_tok_f));
  float*  xv_f  = (float*)(base + alloc(sz_tok_f));
  bf16_t* Qb    = (bf16_t*)(base + alloc(sz_tok_bf));
  bf16_t* Kb    = (bf16_t*)(base + alloc(sz_tok_bf));
  bf16_t* Vtb   = (bf16_t*)(base + alloc(sz_tok_bf));
  float*  ak_f  = (float*)(base + alloc((size_t)ALEN * DIM * sizeof(float)));
  float*  av_f  = (float*)(base + alloc((size_t)ALEN * DIM * sizeof(float)));
  float*  at_f  = (float*)(base + alloc(sz_tok_f));
  bf16_t* at_bf = (bf16_t*)(base + alloc(sz_tok_bf));
  (void)ws_size; (void)in_sizes; (void)n_in; (void)out_size;

  auto cvt = [&](const float* s, bf16_t* d, size_t n) {
    cvt_f32_bf16<<<dim3((unsigned)((n + 255) / 256)), dim3(256), 0, stream>>>(s, d, n);
  };
  // -------- 1) precision conversion
  cvt(x,       x_bf,  (size_t)NTOK * DIM);
  cvt(wq,      wq_bf, (size_t)DIM * DIM);
  cvt(wk,      wk_bf, (size_t)DIM * DIM);
  cvt(wv,      wv_bf, (size_t)DIM * DIM);
  cvt(wo,      wo_bf, (size_t)DIM * DIM);
  cvt(adapter, ad_bf, (size_t)ALEN * DIM);

  auto gemm = [&](const bf16_t* A, const bf16_t* B, float* C, int M, int N, int K) {
    dim3 grid((unsigned)(N / BN), (unsigned)((M + BM - 1) / BM));
    gemm_bf16<<<grid, dim3(256), 0, stream>>>(A, B, C, M, N, K);
  };
  // -------- 2) Q/K/V projections + adapter projections
  gemm(x_bf,  wq_bf, xq_f, NTOK, DIM, DIM);
  gemm(x_bf,  wk_bf, xk_f, NTOK, DIM, DIM);
  gemm(x_bf,  wv_bf, xv_f, NTOK, DIM, DIM);
  gemm(ad_bf, wk_bf, ak_f, ALEN, DIM, DIM);
  gemm(ad_bf, wv_bf, av_f, ALEN, DIM, DIM);

  // -------- 3) RoPE + layouts
  {
    size_t n = (size_t)BSZ * SEQ * NH * (HD / 2);
    rope_qk<<<dim3((unsigned)((n + 255) / 256)), dim3(256), 0, stream>>>(
        xq_f, xk_f, fcos, fsin, Qb, Kb);
    v_transpose<<<dim3(SEQ / 32, HD / 32, BSZ * NH), dim3(32, 8), 0, stream>>>(xv_f, Vtb);
  }

  // -------- 4) flash attention (causal) + adapter attention
  flash_attn<<<dim3(SEQ / FA_BQ, NH, BSZ), dim3(128), 0, stream>>>(Qb, Kb, Vtb, at_f);
  {
    unsigned waves = (unsigned)((size_t)BSZ * NH * SEQ);
    adapter_attn<<<dim3(waves / 8), dim3(256), 0, stream>>>(Qb, ak_f, av_f, gate, at_f);
  }

  // -------- 5) output projection
  cvt(at_f, at_bf, (size_t)NTOK * DIM);
  gemm(at_bf, wo_bf, out, NTOK, DIM, DIM);
}
